// SpatialEncoder3d_1159641170465
// MI455X (gfx1250) — matrix operand order repaired and checked by reading, not compiled, not run on hardware
//
#include <hip/hip_runtime.h>
#include <math.h>

typedef _Float16 v16h __attribute__((ext_vector_type(16)));
typedef _Float16 v8h  __attribute__((ext_vector_type(8)));
typedef _Float16 v2h  __attribute__((ext_vector_type(2)));
typedef float    v8f  __attribute__((ext_vector_type(8)));

#define NK 128   // NUM_KERNELS
#define NH 32    // NUM_HEADS
#define NN 512   // nodes per batch

static __device__ __forceinline__ unsigned pk16(float a, float b) {
    v2h t; t[0] = (_Float16)a; t[1] = (_Float16)b;   // -> v_cvt_pk_f16_f32
    return __builtin_bit_cast(unsigned, t);
}

__global__ __launch_bounds__(256) void spatial_encoder3d_wmma(
    const float* __restrict__ coord,
    const int*   __restrict__ node_type,
    const float* __restrict__ means,
    const float* __restrict__ stds,
    const float* __restrict__ W1,
    const float* __restrict__ b1,
    const float* __restrict__ W2,
    const float* __restrict__ b2,
    const float* __restrict__ gamma_emb,
    const float* __restrict__ beta_emb,
    float* __restrict__ out)
{
    __shared__ __align__(16) _Float16 sW1[NK * NK];  // 32 KB, row-major [j][k]
    __shared__ __align__(16) _Float16 sW2[NH * NK];  //  8 KB, row-major [h][j]
    __shared__ __align__(16) float sA[NK], sB[NK], sC[NK];  // gaussian-in-log form
    __shared__ __align__(16) float sB1[NK], sB2[NH];

    const int tid = threadIdx.x;
    for (int i = tid; i < NK * NK; i += 256) sW1[i] = (_Float16)W1[i];
    for (int i = tid; i < NH * NK; i += 256) sW2[i] = (_Float16)W2[i];
    if (tid < NK) {
        // gk(x) = inv_norm * exp(-0.5*((x-mu)*inv)^2) = exp(A*x^2 + B*x + C)
        const float mu  = means[tid];
        const float inv = 1.0f / (fabsf(stds[tid]) + 0.01f);   // 128 divs per block total
        const float a   = -0.5f * inv * inv;
        sA[tid] = a;
        sB[tid] = -2.0f * a * mu;
        sC[tid] = a * mu * mu + logf(0.3989422804014327f * inv);
        sB1[tid] = b1[tid];
    }
    if (tid < NH) sB2[tid] = b2[tid];
    __syncthreads();

    const int lane = tid & 31;
    const int wid  = tid >> 5;
    const int r    = lane & 15;   // edge row within tile / A-matrix row
    const int hi   = lane >> 4;   // half-wave select
    const int hi8  = hi * 8;

    const int tile = blockIdx.x * 8 + wid;   // 16 edges per wave-tile
    const int e0   = tile * 16;              // base edge (b*N*N + n*N + m0)
    const int b    = e0 >> 18;               // N*N = 262144
    const int n    = (e0 >> 9) & (NN - 1);
    const int m    = (e0 & (NN - 1)) + r;    // N % 16 == 0 -> same (b,n) per tile

    // ---- per-edge scalar prologue: distance + embedding-gated affine ----
    const float* cn = coord + (size_t)(b * NN + n) * 3;
    const float* cm = coord + (size_t)(b * NN + m) * 3;
    const float dx = cn[0] - cm[0], dy = cn[1] - cm[1], dz = cn[2] - cm[2];
    const float dist = sqrtf(dx * dx + dy * dy + dz * dz);

    const int tn = node_type[b * NN + n];
    const int tm = node_type[b * NN + m];
    const float gamma = gamma_emb[tn + 2] + gamma_emb[tm + 103]; // MAX_NODE_TYPE+3
    const float beta  = beta_emb[tn + 2]  + beta_emb[tm + 103];
    const float x  = gamma * dist + beta;
    const float x2 = x * x;

    const v8f zero = {0.f, 0.f, 0.f, 0.f, 0.f, 0.f, 0.f, 0.f};
    v8f hacc[8];   // h^T tiles: lane holds h[r, j = t*16 + hi*8 + v]
#pragma unroll
    for (int t = 0; t < 8; ++t) hacc[t] = zero;

    // ---- GEMM1: h^T = W1_tile x gk^T  (K = 128 in 4 chunks of 32) ----
#pragma unroll
    for (int kc = 0; kc < 4; ++kc) {
        const int kb = kc * 32 + hi * 16;   // this lane's 16 contiguous k's
        v16h gkb;                           // B-operand: gk^T fragment
#pragma unroll
        for (int i = 0; i < 16; i += 4) {
            const float4 av = *(const float4*)(sA + kb + i);
            const float4 bv = *(const float4*)(sB + kb + i);
            const float4 cv = *(const float4*)(sC + kb + i);
            const float aa[4] = {av.x, av.y, av.z, av.w};
            const float bb[4] = {bv.x, bv.y, bv.z, bv.w};
            const float cc[4] = {cv.x, cv.y, cv.z, cv.w};
#pragma unroll
            for (int u = 0; u < 4; ++u) {
                const float e = fmaf(aa[u], x2, fmaf(bb[u], x, cc[u]));
                gkb[i + u] = (_Float16)__expf(e);    // v_exp_f32 path
            }
        }
#pragma unroll
        for (int t = 0; t < 8; ++t) {
            // A-operand: W1 row j = t*16 + r ; K contiguous at +hi*16 (+0..15)
            const v8h* q = (const v8h*)(sW1 + (t * 16 + r) * NK + kc * 32 + hi * 16);
            const v8h alo = q[0], ahi2 = q[1];
            v16h a;
#pragma unroll
            for (int i = 0; i < 8; ++i) { a[i] = alo[i]; a[i + 8] = ahi2[i]; }
            hacc[t] = __builtin_amdgcn_wmma_f32_16x16x32_f16(
                false, a, false, gkb, (short)0, hacc[t], false, false);
        }
    }

    // ---- bias + GELU (tanh form, rcp-based sigmoid; feeds f16 so approx is in-noise) ----
    const float GA = 1.5957691216f;   // 2 * sqrt(2/pi)
    const float GB = 0.0713548162f;   // GA * 0.044715
#pragma unroll
    for (int t = 0; t < 8; ++t) {
        const float4 c0 = *(const float4*)(sB1 + t * 16 + hi8);
        const float4 c1 = *(const float4*)(sB1 + t * 16 + hi8 + 4);
        const float bias[8] = {c0.x, c0.y, c0.z, c0.w, c1.x, c1.y, c1.z, c1.w};
#pragma unroll
        for (int v = 0; v < 8; ++v) {
            const float hv  = hacc[t][v] + bias[v];
            const float arg = hv * fmaf(GB, hv * hv, GA);
            const float ex  = __expf(arg);
            const float sg  = __builtin_amdgcn_rcpf(ex + 1.0f);  // 1/(1+e^arg)
            hacc[t][v] = hv - hv * sg;   // 0.5*hv*(1+tanh(arg/2))
        }
    }

    // ---- GEMM2: out^T = W2_tile x h^T ----
    v8f oacc[2] = {zero, zero};
#pragma unroll
    for (int jc = 0; jc < 4; ++jc) {
        // Pack to f16 pairs first, then half-swap lanes L <-> L+16 (2 values/shuffle)
        const int t0 = 2 * jc, t1 = t0 + 1;
        unsigned own0[4], own1[4], oth0[4], oth1[4];
#pragma unroll
        for (int w = 0; w < 4; ++w) {
            own0[w] = pk16(hacc[t0][2 * w], hacc[t0][2 * w + 1]);
            own1[w] = pk16(hacc[t1][2 * w], hacc[t1][2 * w + 1]);
            oth0[w] = (unsigned)__shfl_xor((int)own0[w], 16, 32);
            oth1[w] = (unsigned)__shfl_xor((int)own1[w], 16, 32);
        }
        union { v16h h; unsigned u[8]; } bf;
#pragma unroll
        for (int w = 0; w < 4; ++w) {
            bf.u[w]     = (hi == 0) ? own0[w] : oth1[w];
            bf.u[w + 4] = (hi == 0) ? oth0[w] : own1[w];
        }
#pragma unroll
        for (int ht = 0; ht < 2; ++ht) {
            const v8h* q = (const v8h*)(sW2 + (ht * 16 + r) * NK + jc * 32 + hi * 16);
            const v8h alo = q[0], ahi2 = q[1];
            v16h a;
#pragma unroll
            for (int i = 0; i < 8; ++i) { a[i] = alo[i]; a[i + 8] = ahi2[i]; }
            oacc[ht] = __builtin_amdgcn_wmma_f32_16x16x32_f16(
                false, a, false, bf.h, (short)0, oacc[ht], false, false);
        }
    }

    // ---- head bias + vectorized store: lane writes out[e0+r, ht*16+hi*8 .. +7] ----
    float* op = out + (size_t)(e0 + r) * NH;
#pragma unroll
    for (int ht = 0; ht < 2; ++ht) {
        const float4 bb0 = *(const float4*)(sB2 + ht * 16 + hi8);
        const float4 bb1 = *(const float4*)(sB2 + ht * 16 + hi8 + 4);
        float4 o0, o1;
        o0.x = oacc[ht][0] + bb0.x; o0.y = oacc[ht][1] + bb0.y;
        o0.z = oacc[ht][2] + bb0.z; o0.w = oacc[ht][3] + bb0.w;
        o1.x = oacc[ht][4] + bb1.x; o1.y = oacc[ht][5] + bb1.y;
        o1.z = oacc[ht][6] + bb1.z; o1.w = oacc[ht][7] + bb1.w;
        *(float4*)(op + ht * 16 + hi8)     = o0;
        *(float4*)(op + ht * 16 + hi8 + 4) = o1;
    }
}

extern "C" void kernel_launch(void* const* d_in, const int* in_sizes, int n_in,
                              void* d_out, int out_size, void* d_ws, size_t ws_size,
                              hipStream_t stream) {
    const float* coord     = (const float*)d_in[0];
    const int*   node_type = (const int*)  d_in[1];
    const float* means     = (const float*)d_in[2];
    const float* stds      = (const float*)d_in[3];
    const float* W1        = (const float*)d_in[4];
    const float* b1        = (const float*)d_in[5];
    const float* W2        = (const float*)d_in[6];
    const float* b2        = (const float*)d_in[7];
    const float* gamma_emb = (const float*)d_in[8];
    const float* beta_emb  = (const float*)d_in[9];
    float* out = (float*)d_out;

    const long long edges = (long long)out_size / NH;  // b*N*N = 524288
    const int blocks = (int)(edges / 128);             // 16 edges/wave * 8 waves/block
    spatial_encoder3d_wmma<<<blocks, 256, 0, stream>>>(
        coord, node_type, means, stds, W1, b1, W2, b2, gamma_emb, beta_emb, out);
}
